// SoAP_27590869909756
// MI455X (gfx1250) — matrix operand order repaired and checked by reading, not compiled, not run on hardware
//
#include <hip/hip_runtime.h>
#include <hip/hip_bf16.h>
#include <math.h>

#define SOAP_EPS 1e-8f
#define SOAP_B   64
#define SOAP_N   16384
#define SOAP_D   64
#define SOAP_OUT 256

typedef _Float16 v16h __attribute__((ext_vector_type(16)));
typedef _Float16 v8h  __attribute__((ext_vector_type(8)));
typedef float    v8f  __attribute__((ext_vector_type(8)));
typedef float    f4v  __attribute__((ext_vector_type(4)));

// ---------------------------------------------------------------------------
// Stage 1: cov[b] = X[b]^T X[b] / N  via split-fp16 WMMA (hi*hi + hi*lo + lo*hi)
// grid = 64 (batch), block = 512 (16 waves). Each wave owns one 16x16 tile.
// LDS holds chunk transposed [d][k], k-chunk = 32, f16 hi/lo planes.
// ---------------------------------------------------------------------------
#define STR 40  // halfs per LDS row (32 data + 8 pad) -> 80B stride, 16B aligned

__global__ void soap_cov_wmma(const float* __restrict__ x,
                              float* __restrict__ cov) {
    __shared__ _Float16 shT_hi[SOAP_D * STR];
    __shared__ _Float16 shT_lo[SOAP_D * STR];

    const int b    = blockIdx.x;
    const int tid  = threadIdx.x;
    const int lane = tid & 31;
    const int wave = tid >> 5;          // 0..15
    const int ti   = wave >> 2;         // tile row  (i-block)
    const int tj   = wave & 3;          // tile col  (j-block)
    const int m    = lane & 15;
    const int hs   = lane >> 4;         // lane half-select

    const float* xb = x + (size_t)b * SOAP_N * SOAP_D;

    // this thread's global-load slot: one float4 of the 32x64 chunk
    const int ld_k = tid >> 4;          // 0..31  (chunk row, n-index)
    const int ld_d = (tid & 15) * 4;    // 0..60  (d-index, 4 at a time)

    v8f acc = {};

    for (int n0 = 0; n0 < SOAP_N; n0 += 32) {
        // ---- load chunk, split into f16 hi/lo, store transposed ----
        f4v v = *(const f4v*)(xb + (size_t)(n0 + ld_k) * SOAP_D + ld_d);
        if (n0 + 32 < SOAP_N)
            __builtin_prefetch(xb + (size_t)(n0 + 32 + ld_k) * SOAP_D + ld_d, 0, 1);

        _Float16 hi[4], lo[4];
#pragma unroll
        for (int j = 0; j < 4; ++j) {
            float f = fmaxf(v[j], SOAP_EPS);
            _Float16 h = (_Float16)f;
            hi[j] = h;
            lo[j] = (_Float16)(f - (float)h);
        }
        __syncthreads();                 // prior iteration's reads complete
#pragma unroll
        for (int j = 0; j < 4; ++j) {
            shT_hi[(ld_d + j) * STR + ld_k] = hi[j];
            shT_lo[(ld_d + j) * STR + ld_k] = lo[j];
        }
        __syncthreads();                 // chunk visible to all waves

        // ---- build fragments per CDNA5 16-bit A 16x32 / B 32x16 layouts ----
        // A: element e in [0,8): K = hs*8 + e ; e in [8,16): K = 16 + hs*8 + e-8
        const _Float16* rAh = shT_hi + (ti * 16 + m) * STR;
        const _Float16* rAl = shT_lo + (ti * 16 + m) * STR;
        v8h ah0 = *(const v8h*)(rAh + hs * 8);
        v8h ah1 = *(const v8h*)(rAh + 16 + hs * 8);
        v8h al0 = *(const v8h*)(rAl + hs * 8);
        v8h al1 = *(const v8h*)(rAl + 16 + hs * 8);
        v16h a_hi = __builtin_shufflevector(ah0, ah1, 0,1,2,3,4,5,6,7,8,9,10,11,12,13,14,15);
        v16h a_lo = __builtin_shufflevector(al0, al1, 0,1,2,3,4,5,6,7,8,9,10,11,12,13,14,15);

        // B: n = lane&15 ; element e in [0,8): K = hs*8 + e ; e in [8,16): K = 16 + hs*8 + e-8
        const _Float16* rBh = shT_hi + (tj * 16 + m) * STR;
        const _Float16* rBl = shT_lo + (tj * 16 + m) * STR;
        v8h bh0 = *(const v8h*)(rBh + hs * 8);
        v8h bh1 = *(const v8h*)(rBh + 16 + hs * 8);
        v8h bl0 = *(const v8h*)(rBl + hs * 8);
        v8h bl1 = *(const v8h*)(rBl + 16 + hs * 8);
        v16h b_hi = __builtin_shufflevector(bh0, bh1, 0,1,2,3,4,5,6,7,8,9,10,11,12,13,14,15);
        v16h b_lo = __builtin_shufflevector(bl0, bl1, 0,1,2,3,4,5,6,7,8,9,10,11,12,13,14,15);

        // ---- split-precision product: hi*hi + hi*lo + lo*hi (f32 accum) ----
        acc = __builtin_amdgcn_wmma_f32_16x16x32_f16(false, a_hi, false, b_hi,
                                                     (short)0, acc, false, false);
        acc = __builtin_amdgcn_wmma_f32_16x16x32_f16(false, a_hi, false, b_lo,
                                                     (short)0, acc, false, false);
        acc = __builtin_amdgcn_wmma_f32_16x16x32_f16(false, a_lo, false, b_hi,
                                                     (short)0, acc, false, false);
    }

    // ---- store 16x16 tile: C layout -> M = r + 8*hs, Ncol = lane&15 ----
    const float invN = 1.0f / (float)SOAP_N;
    const int col     = tj * 16 + m;
    const int rowbase = ti * 16 + 8 * hs;
    float* covb = cov + (size_t)b * SOAP_D * SOAP_D;
#pragma unroll
    for (int r = 0; r < 8; ++r)
        covb[(rowbase + r) * SOAP_D + col] = acc[r] * invN;
}

// ---------------------------------------------------------------------------
// Stage 2: symmetric eigendecomposition (parallel cyclic Jacobi), matrix log,
// power normalization.  grid = 64 (batch), block = 256.
// ---------------------------------------------------------------------------
#define JAC_SWEEPS 10
#define APAD 65

__global__ void soap_jacobi_logm(const float* __restrict__ cov,
                                 const float* __restrict__ p,
                                 float* __restrict__ m_out) {
    __shared__ float A[SOAP_D * APAD];
    __shared__ float Vm[SOAP_D * APAD];
    __shared__ float cS[32], sS[32];
    __shared__ int   pS[32], qS[32];
    __shared__ float lam[SOAP_D];

    const int b   = blockIdx.x;
    const int tid = threadIdx.x;

    for (int idx = tid; idx < SOAP_D * SOAP_D; idx += 256) {
        int i = idx >> 6, j = idx & 63;
        A[i * APAD + j]  = cov[(size_t)b * SOAP_D * SOAP_D + idx];
        Vm[i * APAD + j] = (i == j) ? 1.0f : 0.0f;
    }
    __syncthreads();

    for (int sweep = 0; sweep < JAC_SWEEPS; ++sweep) {
        for (int r = 0; r < 63; ++r) {
            // 32 disjoint pairs (round-robin tournament ordering)
            if (tid < 32) {
                int P, Q;
                if (tid == 0) { P = 63; Q = r % 63; }
                else {
                    P = (r + tid) % 63;
                    Q = (r + 63 - tid) % 63;
                }
                float app = A[P * APAD + P];
                float aqq = A[Q * APAD + Q];
                float apq = A[P * APAD + Q];
                float c = 1.0f, s = 0.0f;
                if (fabsf(apq) > 1e-36f) {
                    float theta = 0.5f * (aqq - app) / apq;
                    float t = (theta >= 0.0f ? 1.0f : -1.0f) /
                              (fabsf(theta) + sqrtf(theta * theta + 1.0f));
                    c = rsqrtf(1.0f + t * t);
                    s = t * c;
                }
                cS[tid] = c; sS[tid] = s; pS[tid] = P; qS[tid] = Q;
            }
            __syncthreads();
            // column pass: A <- A*G, V <- V*G   (2048 (row,pair) tasks)
            for (int t = tid; t < 2048; t += 256) {
                int k = t & 31, rr = t >> 5;
                int P = pS[k], Q = qS[k];
                float c = cS[k], s = sS[k];
                float ap = A[rr * APAD + P], aq = A[rr * APAD + Q];
                A[rr * APAD + P] = c * ap - s * aq;
                A[rr * APAD + Q] = s * ap + c * aq;
                float vp = Vm[rr * APAD + P], vq = Vm[rr * APAD + Q];
                Vm[rr * APAD + P] = c * vp - s * vq;
                Vm[rr * APAD + Q] = s * vp + c * vq;
            }
            __syncthreads();
            // row pass: A <- G^T * A
            for (int t = tid; t < 2048; t += 256) {
                int k = t & 31, cc = t >> 5;
                int P = pS[k], Q = qS[k];
                float c = cS[k], s = sS[k];
                float ap = A[P * APAD + cc], aq = A[Q * APAD + cc];
                A[P * APAD + cc] = c * ap - s * aq;
                A[Q * APAD + cc] = s * ap + c * aq;
            }
            __syncthreads();
        }
    }

    if (tid < SOAP_D)
        lam[tid] = logf(fmaxf(A[tid * APAD + tid], SOAP_EPS));
    __syncthreads();

    const float pw = p[0];
    for (int idx = tid; idx < SOAP_D * SOAP_D; idx += 256) {
        int i = idx >> 6, j = idx & 63;
        float sum = 0.0f;
#pragma unroll 8
        for (int k = 0; k < SOAP_D; ++k)
            sum += Vm[i * APAD + k] * lam[k] * Vm[j * APAD + k];
        // m = max(m, eps); sign(m)*|m|^p == m^p since m > 0; max again (parity)
        sum = fmaxf(sum, SOAP_EPS);
        sum = powf(sum, pw);
        sum = fmaxf(sum, SOAP_EPS);
        m_out[(size_t)b * SOAP_D * SOAP_D + idx] = sum;
    }
}

// ---------------------------------------------------------------------------
// Stage 3: FC [4096 -> 256] + row L2 normalize. grid = 64, block = 256.
// fc_w (4 MB) stays resident in L2 and is reused by all 64 blocks.
// ---------------------------------------------------------------------------
__global__ void soap_fc_norm(const float* __restrict__ m_in,
                             const float* __restrict__ fc_w,
                             const float* __restrict__ fc_b,
                             float* __restrict__ out) {
    __shared__ float feat[SOAP_D * SOAP_D];
    __shared__ float red[256];

    const int b   = blockIdx.x;
    const int tid = threadIdx.x;

    for (int k = tid; k < SOAP_D * SOAP_D; k += 256)
        feat[k] = m_in[(size_t)b * SOAP_D * SOAP_D + k];
    __syncthreads();

    float acc = fc_b[tid];
    const f4v* wrow = (const f4v*)(fc_w + (size_t)tid * SOAP_D * SOAP_D);
#pragma unroll 4
    for (int k4 = 0; k4 < (SOAP_D * SOAP_D) / 4; ++k4) {
        f4v w = wrow[k4];
        int k = k4 * 4;
        acc += feat[k] * w[0] + feat[k + 1] * w[1] +
               feat[k + 2] * w[2] + feat[k + 3] * w[3];
    }

    red[tid] = acc * acc;
    __syncthreads();
    for (int s = 128; s > 0; s >>= 1) {
        if (tid < s) red[tid] += red[tid + s];
        __syncthreads();
    }
    float norm = sqrtf(red[0]);
    out[(size_t)b * SOAP_OUT + tid] = acc / fmaxf(norm, 1e-12f);
}

// ---------------------------------------------------------------------------
extern "C" void kernel_launch(void* const* d_in, const int* in_sizes, int n_in,
                              void* d_out, int out_size, void* d_ws, size_t ws_size,
                              hipStream_t stream) {
    const float* x    = (const float*)d_in[0];   // [64,16384,64]
    const float* fc_w = (const float*)d_in[1];   // [256,4096]
    const float* fc_b = (const float*)d_in[2];   // [256]
    const float* p    = (const float*)d_in[3];   // [1]
    float* out = (float*)d_out;                  // [64,256]

    float* cov  = (float*)d_ws;                         // 64*64*64 floats (1 MB)
    float* mbuf = cov + (size_t)SOAP_B * SOAP_D * SOAP_D; // 64*4096 floats (1 MB)

    soap_cov_wmma  <<<SOAP_B, 512, 0, stream>>>(x, cov);
    soap_jacobi_logm<<<SOAP_B, 256, 0, stream>>>(cov, p, mbuf);
    soap_fc_norm   <<<SOAP_B, 256, 0, stream>>>(mbuf, fc_w, fc_b, out);
    (void)in_sizes; (void)n_in; (void)out_size; (void)ws_size;
}
